// StaticSparseGAT_44169443672636
// MI455X (gfx1250) — matrix operand order repaired and checked by reading, not compiled, not run on hardware
//
#include <hip/hip_runtime.h>

#define NNODES 50000
#define NEDGES 800000
#define DIM 128
#define NHEADS 8

typedef __attribute__((ext_vector_type(16))) __bf16 bf16x16;
typedef __attribute__((ext_vector_type(8)))  float  v8f;

// ---------- CDNA5 async global->LDS (guarded; sync fallback keeps build safe) ----
#if defined(__HIP_DEVICE_COMPILE__) && \
    __has_builtin(__builtin_amdgcn_global_load_async_to_lds_b128)
#define HAVE_ASYNC_LDS 1
typedef int v4i_vs __attribute__((__vector_size__(16)));
typedef __attribute__((address_space(1))) v4i_vs as1_v4i;  // global
typedef __attribute__((address_space(3))) v4i_vs as3_v4i;  // LDS
__device__ __forceinline__ void async_g2l_b128(const void* g, void* l) {
    // low 32 bits of a generic LDS pointer are the LDS byte offset (ISA 10.2)
    __builtin_amdgcn_global_load_async_to_lds_b128(
        (as1_v4i*)(size_t)g, (as3_v4i*)(unsigned)(size_t)l, 0, 0);
}
__device__ __forceinline__ void async_wait0() {
#if __has_builtin(__builtin_amdgcn_s_wait_asynccnt)
    __builtin_amdgcn_s_wait_asynccnt(0);
#else
    asm volatile("s_wait_asynccnt 0x0" ::: "memory");
#endif
}
#endif

// ---------------- helpers ----------------
__device__ inline unsigned enc_f32(float f) {
    unsigned u = __float_as_uint(f);
    return (u & 0x80000000u) ? ~u : (u | 0x80000000u);
}
__device__ inline float dec_f32(unsigned u) {
    unsigned v = (u & 0x80000000u) ? (u & 0x7FFFFFFFu) : ~u;
    return __uint_as_float(v);
}

// ---------------- weight repack: f32 row-major -> bf16 WMMA B-fragment layout ----
// Fragment (ntile 0..7, ktile 0..3): 32 lanes x 16 bf16, contiguous per lane.
// Element e of lane L: B[K,N] with K = ktile*32 + (L/16)*8 + (e<8 ? e : e+8), N = ntile*16 + (L%16)
// and B = W.T  =>  value = W[N_global * 128 + K_global].
__global__ void prep_weights(const float* __restrict__ W1, const float* __restrict__ W2,
                             const float* __restrict__ Wv, const float* __restrict__ Wres,
                             const float* __restrict__ Wout, __bf16* __restrict__ wprep) {
    int t = blockIdx.x * blockDim.x + threadIdx.x;
    if (t >= 5 * 8 * 4 * 32) return;
    int lane = t & 31;
    int kt   = (t >> 5) & 3;
    int nt   = (t >> 7) & 7;
    int m    = t >> 10;
    const float* W = (m == 0) ? W1 : (m == 1) ? W2 : (m == 2) ? Wv : (m == 3) ? Wres : Wout;
    int half = lane >> 4, lr = lane & 15;
    int Kg = kt * 32 + half * 8;
    __bf16* dst = wprep + (size_t)m * 16384 + ((size_t)(nt * 4 + kt) * 32 + lane) * 16;
#pragma unroll
    for (int e = 0; e < 16; e++)
        dst[e] = (__bf16)W[(size_t)(nt * 16 + lr) * DIM + Kg + (e < 8 ? e : e + 8)];
}

__global__ void zero_f32(float* __restrict__ p, int n) {
    int i = blockIdx.x * blockDim.x + threadIdx.x;
    if (i < n) p[i] = 0.0f;
}

// Build one 16x32 bf16 A-fragment from an f32 LDS tile (stride 132 floats).
__device__ __forceinline__ bf16x16 build_afrag(const float* tilef, int row,
                                               int kk, int half) {
    const float* base = tilef + row * 132 + kk * 32 + half * 8;
    bf16x16 a;
#pragma unroll
    for (int e = 0; e < 8; e++) {
        a[e]     = (__bf16)base[e];
        a[8 + e] = (__bf16)base[16 + e];
    }
    return a;
}

// Stage an MROWSx128 f32 tile into LDS (async if available).
template <int MROWS>
__device__ __forceinline__ void stage_tile(const float* __restrict__ src,
                                           float* tilef, int tid) {
    constexpr int SLOTS = MROWS * 32;  // 4 floats per slot
#pragma unroll
    for (int i = 0; i < SLOTS / 256; i++) {
        int slot = tid + 256 * i;
        int row = slot >> 5, c0 = (slot & 31) * 4;
#if defined(HAVE_ASYNC_LDS)
        async_g2l_b128(src + (size_t)row * DIM + c0, tilef + row * 132 + c0);
#else
        *(float4*)(tilef + row * 132 + c0) =
            *(const float4*)(src + (size_t)row * DIM + c0);
#endif
    }
#if defined(HAVE_ASYNC_LDS)
    async_wait0();
#endif
    __syncthreads();
}

// ---------------- node GEMMs: H1 = H W1^T, H2 = H W2^T, V = H Wv^T, R = H Wres^T + res_b
// 80 rows per block (N = 625*80): B-fragments cached in VGPRs, reused over 5 m-tiles.
__global__ void __launch_bounds__(256)
gemm_node(const float* __restrict__ H, const __bf16* __restrict__ wprep,
          const float* __restrict__ res_b,
          float* __restrict__ H1, float* __restrict__ H2,
          float* __restrict__ V, float* __restrict__ R) {
    __shared__ __align__(16) float tilef[80 * 132];
    const int tid = threadIdx.x;
    const int rowBase = blockIdx.x * 80;
    stage_tile<80>(H + (size_t)rowBase * DIM, tilef, tid);

    const int w = tid >> 5, lane = tid & 31, half = lane >> 4, lr = lane & 15;
    const int col = w * 16 + lr;

    float* outs[4] = {H1, H2, V, R};
#pragma unroll
    for (int m = 0; m < 4; m++) {
        bf16x16 bfrag[4];
#pragma unroll
        for (int kk = 0; kk < 4; kk++)
            bfrag[kk] = *(const bf16x16*)(wprep + (size_t)m * 16384 +
                                          ((size_t)(w * 4 + kk) * 32 + lane) * 16);
        float bias = (m == 3) ? res_b[col] : 0.0f;
        float* out = outs[m];
#pragma unroll
        for (int mt = 0; mt < 5; mt++) {
            v8f acc = {0.f, 0.f, 0.f, 0.f, 0.f, 0.f, 0.f, 0.f};
#pragma unroll
            for (int kk = 0; kk < 4; kk++) {
                bf16x16 a = build_afrag(tilef, mt * 16 + lr, kk, half);
                acc = __builtin_amdgcn_wmma_f32_16x16x32_bf16(
                    false, a, false, bfrag[kk], (short)0, acc, false, false);
            }
#pragma unroll
            for (int r = 0; r < 8; r++)
                out[(size_t)(rowBase + mt * 16 + r + 8 * half) * DIM + col] =
                    acc[r] + bias;
        }
    }
}

// ---------------- edge pass 1: logits (leaky-relu) + segment max (monotonic uint atomicMax)
__global__ void __launch_bounds__(256)
edge_logits(const float* __restrict__ H1, const float* __restrict__ H2,
            const long long* __restrict__ ei, const float* __restrict__ P,
            const float* __restrict__ det, const float* __restrict__ W3,
            const float* __restrict__ W4, float* __restrict__ logits,
            unsigned* __restrict__ mxenc) {
    const int w = threadIdx.x >> 5, lane = threadIdx.x & 31;
    const int e = blockIdx.x * 8 + w;
    if (e >= NEDGES) return;
    const int src = (int)ei[e];
    const int dst = (int)ei[(size_t)NEDGES + e];
    const float4 h1 = *(const float4*)(H1 + (size_t)dst * DIM + lane * 4);
    const float4 h2 = *(const float4*)(H2 + (size_t)src * DIM + lane * 4);
    const float4 w3 = *(const float4*)(W3 + lane * 4);
    const float p = P[e];
    float4 eb;
    eb.x = h1.x + h2.x + p * w3.x;
    eb.y = h1.y + h2.y + p * w3.y;
    eb.z = h1.z + h2.z + p * w3.z;
    eb.w = h1.w + h2.w + p * w3.w;
    float acc[NHEADS];
#pragma unroll
    for (int h = 0; h < NHEADS; h++) {
        const float4 w4 = *(const float4*)(W4 + h * DIM + lane * 4);
        acc[h] = eb.x * w4.x + eb.y * w4.y + eb.z * w4.z + eb.w * w4.w;
    }
#pragma unroll
    for (int off = 16; off; off >>= 1) {
#pragma unroll
        for (int h = 0; h < NHEADS; h++) acc[h] += __shfl_xor(acc[h], off, 32);
    }
    if (lane == 0) {
        const float d = det[e];
#pragma unroll
        for (int h = 0; h < NHEADS; h++) {
            float lg = acc[h] + d;
            lg = (lg >= 0.0f) ? lg : 0.2f * lg;
            logits[(size_t)e * NHEADS + h] = lg;
            atomicMax(&mxenc[(size_t)dst * NHEADS + h], enc_f32(lg));
        }
    }
}

// ---------------- edge pass 2: exp(logit - max) in place + segment sum
__global__ void __launch_bounds__(256)
edge_exp(float* __restrict__ logits, const long long* __restrict__ ei,
         const unsigned* __restrict__ mxenc, float* __restrict__ denom) {
    int idx = blockIdx.x * blockDim.x + threadIdx.x;
    if (idx >= NEDGES * NHEADS) return;
    int e = idx >> 3, h = idx & 7;
    int dst = (int)ei[(size_t)NEDGES + e];
    float m = dec_f32(mxenc[(size_t)dst * NHEADS + h]);
    float ex = __expf(logits[idx] - m);
    logits[idx] = ex;
    atomicAdd(&denom[(size_t)dst * NHEADS + h], ex);
}

// ---------------- edge pass 3: alpha * V[src] scatter-add into agg (L2-resident)
__global__ void __launch_bounds__(256)
edge_agg(const float* __restrict__ exps, const long long* __restrict__ ei,
         const float* __restrict__ denom, const float* __restrict__ V,
         float* __restrict__ agg) {
    const int w = threadIdx.x >> 5, lane = threadIdx.x & 31;
    const int e = blockIdx.x * 8 + w;
    if (e >= NEDGES) return;
    const int src = (int)ei[e];
    const int dst = (int)ei[(size_t)NEDGES + e];
    const int h = lane >> 2;
    const float a = exps[(size_t)e * NHEADS + h] /
                    (denom[(size_t)dst * NHEADS + h] + 1e-12f);
    const float4 v = *(const float4*)(V + (size_t)src * DIM + lane * 4);
    float* base = agg + (size_t)dst * DIM + lane * 4;
    atomicAdd(base + 0, a * v.x);
    atomicAdd(base + 1, a * v.y);
    atomicAdd(base + 2, a * v.z);
    atomicAdd(base + 3, a * v.w);
}

// ---------------- output GEMM (agg @ Wout^T + b + R) fused with LayerNorm -------
__global__ void __launch_bounds__(256)
gemm_out_ln(const float* __restrict__ AGG, const __bf16* __restrict__ wprep,
            const float* __restrict__ Wout_b, const float* __restrict__ R,
            const float* __restrict__ ln_g, const float* __restrict__ ln_b,
            float* __restrict__ out) {
    __shared__ __align__(16) float tilef[16 * 132];
    __shared__ float ot[16 * 136];
    const int tid = threadIdx.x;
    const int rowBase = blockIdx.x * 16;
    stage_tile<16>(AGG + (size_t)rowBase * DIM, tilef, tid);

    const int w = tid >> 5, lane = tid & 31, half = lane >> 4, lr = lane & 15;
    v8f acc = {0.f, 0.f, 0.f, 0.f, 0.f, 0.f, 0.f, 0.f};
#pragma unroll
    for (int kk = 0; kk < 4; kk++) {
        const bf16x16 b =
            *(const bf16x16*)(wprep + (size_t)4 * 16384 +
                              ((size_t)(w * 4 + kk) * 32 + lane) * 16);
        bf16x16 a = build_afrag(tilef, lr, kk, half);
        acc = __builtin_amdgcn_wmma_f32_16x16x32_bf16(
            false, a, false, b, (short)0, acc, false, false);
    }
    const int col = w * 16 + lr;
    const float bias = Wout_b[col];
#pragma unroll
    for (int r = 0; r < 8; r++) {
        int row = r + 8 * half;
        ot[row * 136 + col] =
            acc[r] + bias + R[(size_t)(rowBase + row) * DIM + col];
    }
    __syncthreads();

    // LayerNorm: wave w handles rows 2w, 2w+1 (lane covers 4 columns)
#pragma unroll
    for (int rr = 0; rr < 2; rr++) {
        int row = w * 2 + rr;
        float x[4], s = 0.f, sq = 0.f;
#pragma unroll
        for (int j = 0; j < 4; j++) {
            x[j] = ot[row * 136 + lane * 4 + j];
            s += x[j];
            sq += x[j] * x[j];
        }
#pragma unroll
        for (int off = 16; off; off >>= 1) {
            s  += __shfl_xor(s,  off, 32);
            sq += __shfl_xor(sq, off, 32);
        }
        float mu  = s * (1.0f / DIM);
        float var = fmaxf(sq * (1.0f / DIM) - mu * mu, 0.0f);
        float rstd = rsqrtf(var + 1e-5f);
#pragma unroll
        for (int j = 0; j < 4; j++) {
            int c = lane * 4 + j;
            out[(size_t)(rowBase + row) * DIM + c] =
                (x[j] - mu) * rstd * ln_g[c] + ln_b[c];
        }
    }
}

// ---------------- launch ----------------
extern "C" void kernel_launch(void* const* d_in, const int* in_sizes, int n_in,
                              void* d_out, int out_size, void* d_ws, size_t ws_size,
                              hipStream_t stream) {
    (void)in_sizes; (void)n_in; (void)out_size; (void)ws_size;
    const float*     H      = (const float*)d_in[0];
    const long long* ei     = (const long long*)d_in[1];
    const float*     P      = (const float*)d_in[2];
    const float*     det    = (const float*)d_in[3];
    const float*     W1     = (const float*)d_in[4];
    const float*     W2     = (const float*)d_in[5];
    const float*     W3     = (const float*)d_in[6];
    const float*     W4     = (const float*)d_in[7];
    const float*     Wv     = (const float*)d_in[8];
    const float*     Wout_w = (const float*)d_in[9];
    const float*     Wout_b = (const float*)d_in[10];
    const float*     res_w  = (const float*)d_in[11];
    const float*     res_b  = (const float*)d_in[12];
    const float*     ln_g   = (const float*)d_in[13];
    const float*     ln_b   = (const float*)d_in[14];
    float* out = (float*)d_out;

    const size_t ND = (size_t)NNODES * DIM;       // 6,400,000
    float* ws = (float*)d_ws;
    float*    H1    = ws + 0 * ND;
    float*    H2    = ws + 1 * ND;
    float*    Vbuf  = ws + 2 * ND;
    float*    Rbuf  = ws + 3 * ND;
    float*    AGG   = ws + 4 * ND;
    float*    LOGIT = ws + 5 * ND;                // E*NH == ND floats
    unsigned* MX    = (unsigned*)(ws + 6 * ND);   // N*NH u32
    float*    DEN   = ws + 6 * ND + (size_t)NNODES * NHEADS;
    __bf16*   WPREP = (__bf16*)(ws + 6 * ND + 2 * (size_t)NNODES * NHEADS); // 32B-aligned

    hipLaunchKernelGGL(prep_weights, dim3(20), dim3(256), 0, stream,
                       W1, W2, Wv, res_w, Wout_w, WPREP);
    // zero agg + mx + denom
    hipLaunchKernelGGL(zero_f32, dim3((unsigned)(ND / 256)), dim3(256), 0, stream,
                       AGG, (int)ND);
    hipLaunchKernelGGL(zero_f32, dim3((NNODES * NHEADS * 2 + 255) / 256), dim3(256), 0, stream,
                       (float*)MX, NNODES * NHEADS * 2);

    hipLaunchKernelGGL(gemm_node, dim3(NNODES / 80), dim3(256), 0, stream,
                       H, WPREP, res_b, H1, H2, Vbuf, Rbuf);

    hipLaunchKernelGGL(edge_logits, dim3((NEDGES + 7) / 8), dim3(256), 0, stream,
                       H1, H2, ei, P, det, W3, W4, LOGIT, MX);

    hipLaunchKernelGGL(edge_exp, dim3((NEDGES * NHEADS + 255) / 256), dim3(256), 0, stream,
                       LOGIT, ei, MX, DEN);

    hipLaunchKernelGGL(edge_agg, dim3((NEDGES + 7) / 8), dim3(256), 0, stream,
                       LOGIT, ei, DEN, Vbuf, AGG);

    hipLaunchKernelGGL(gemm_out_ln, dim3(NNODES / 16), dim3(256), 0, stream,
                       AGG, WPREP, Wout_b, Rbuf, ln_g, ln_b, out);
}
